// MyModel_63642825392703
// MI455X (gfx1250) — compile-verified, matrix operand order checked by reading, not run on hardware
//
#include <hip/hip_runtime.h>

// ---------------------------------------------------------------------------
// MI455X (gfx1250, wave32) fused MLP + QUBO evaluation.
//   h = sigmoid(h @ W^T + b)  x3   -> sample hb = (u < h)
//   out[b] = hb[b] @ Q @ hb[b]     (Q symmetric => hb @ Q == hb @ Q^T)
// All GEMMs are (32 x 1024) @ (1024 x 1024)^T done with v_wmma_f32_16x16x32_bf16.
// ---------------------------------------------------------------------------

typedef __bf16 bf16_t;
typedef bf16_t v16bf __attribute__((ext_vector_type(16)));
typedef float  v8f   __attribute__((ext_vector_type(8)));
typedef float  v4f   __attribute__((ext_vector_type(4)));

constexpr int kN = 1024;   // feature dim (= K = output dim)
constexpr int kB = 32;     // batch rows

// Pack one lane's 16 bf16 A/B elements for V_WMMA_F32_16X16X32_BF16.
// CDNA5 16-bit operand layout (ISA 7.12.2): lane half 0 holds K = base+{0..7},
// base+{16..23}; lane half 1 holds K = base+{8..15}, base+{24..31}.  The
// caller pre-offsets `p` by half*8, so this lane reads runs at +0 and +16.
__device__ __forceinline__ v16bf pack_bf16(const float* __restrict__ p) {
  v4f f0 = *(const v4f*)(p + 0);
  v4f f1 = *(const v4f*)(p + 4);
  v4f f2 = *(const v4f*)(p + 16);
  v4f f3 = *(const v4f*)(p + 20);
  v16bf r;
#pragma unroll
  for (int i = 0; i < 4; ++i) {
    r[i + 0]  = (bf16_t)f0[i];
    r[i + 4]  = (bf16_t)f1[i];
    r[i + 8]  = (bf16_t)f2[i];
    r[i + 12] = (bf16_t)f3[i];
  }
  return r;
}

// One 16x16 output tile per block; 8 waves split K=1024 (128 each, 4 WMMA
// steps), partials reduced through LDS.
//   mode 0: sigmoid(acc + bias)
//   mode 1: sigmoid(acc + bias), then Bernoulli sample vs u (straight-through)
//   mode 2: plain accumulate (QUBO T = hb @ Q^T)
__global__ __launch_bounds__(256) void fused_layer_kernel(
    const float* __restrict__ Hin,   // kB x kN
    const float* __restrict__ W,     // kN x kN, row n holds weights over K
    const float* __restrict__ bias,  // kN (mode 0/1)
    const float* __restrict__ u,     // kB x kN (mode 1)
    float* __restrict__ Hout,        // kB x kN
    int mode) {
  __shared__ float lds[8 * 256];

  const int tid   = threadIdx.x;
  const int wave  = tid >> 5;
  const int lane  = tid & 31;
  const int hsel  = lane >> 4;   // which half of the wave
  const int row16 = lane & 15;

  const int tile   = blockIdx.x;   // 0..127
  const int m_tile = tile & 1;     // batch 32 -> two M tiles
  const int n_tile = tile >> 1;    // 64 N tiles

  const int m = m_tile * 16 + row16;           // A row this lane feeds
  const int n = n_tile * 16 + row16;           // W row this lane feeds (B = W^T)

  const float* __restrict__ arow = Hin + (size_t)m * kN + hsel * 8;
  const float* __restrict__ brow = W   + (size_t)n * kN + hsel * 8;

  v8f acc = {};
  const int k0 = wave * 128;
#pragma unroll
  for (int kk = 0; kk < 4; ++kk) {
    const int k = k0 + kk * 32;
    v16bf a = pack_bf16(arow + k);
    v16bf b = pack_bf16(brow + k);
    // (neg_a, A, neg_b, B, c_mod, C, reuse_a, reuse_b)
    acc = __builtin_amdgcn_wmma_f32_16x16x32_bf16(
        false, a, false, b, (short)0, acc, false, false);
  }

  // Stash this wave's partial tile; position = lane*8 + r.
#pragma unroll
  for (int r = 0; r < 8; ++r) lds[wave * 256 + lane * 8 + r] = acc[r];
  __syncthreads();

  // Thread t owns tile position (lane_p = t>>3, r_p = t&7): sum 8 K-slices.
  float s = 0.f;
#pragma unroll
  for (int w = 0; w < 8; ++w) s += lds[w * 256 + tid];

  const int lane_p  = tid >> 3;
  const int r_p     = tid & 7;
  const int half_p  = lane_p >> 4;
  const int row16_p = lane_p & 15;
  // C/D layout (ISA 7.12.2): VGPR r, lanes 0-15 -> M=r, lanes 16-31 -> M=r+8.
  const int m_out = m_tile * 16 + half_p * 8 + r_p;
  const int n_out = n_tile * 16 + row16_p;

  if (mode != 2) {
    s += bias[n_out];
    s = 1.f / (1.f + __expf(-s));                       // sigmoid
    if (mode == 1)                                      // Bernoulli sample
      s = (u[(size_t)m_out * kN + n_out] < s) ? 1.f : 0.f;
  }
  Hout[(size_t)m_out * kN + n_out] = s;
}

// out[b] = sum_n T[b,n] * Hb[b,n]   (one block per batch row)
__global__ __launch_bounds__(256) void qubo_reduce_kernel(
    const float* __restrict__ T, const float* __restrict__ Hb,
    float* __restrict__ out) {
  __shared__ float lds[256];
  const int b = blockIdx.x;
  const int t = threadIdx.x;
  float s = 0.f;
  for (int i = t; i < kN; i += 256)
    s += T[(size_t)b * kN + i] * Hb[(size_t)b * kN + i];
  lds[t] = s;
  __syncthreads();
#pragma unroll
  for (int off = 128; off > 0; off >>= 1) {
    if (t < off) lds[t] += lds[t + off];
    __syncthreads();
  }
  if (t == 0) out[b] = lds[0];
}

extern "C" void kernel_launch(void* const* d_in, const int* in_sizes, int n_in,
                              void* d_out, int out_size, void* d_ws,
                              size_t ws_size, hipStream_t stream) {
  const float* x  = (const float*)d_in[0];
  const float* u  = (const float*)d_in[1];
  const float* W1 = (const float*)d_in[2];
  const float* b1 = (const float*)d_in[3];
  const float* W2 = (const float*)d_in[4];
  const float* b2 = (const float*)d_in[5];
  const float* W3 = (const float*)d_in[6];
  const float* b3 = (const float*)d_in[7];
  const float* Q  = (const float*)d_in[8];
  float* out = (float*)d_out;

  float* H1 = (float*)d_ws;            // 32x1024 activations (f32)
  float* H2 = H1 + (size_t)kB * kN;
  float* Hb = H2 + (size_t)kB * kN;    // sampled binary vector
  float* T  = Hb + (size_t)kB * kN;    // hb @ Q^T

  fused_layer_kernel<<<128, 256, 0, stream>>>(x,  W1, b1, nullptr, H1, 0);
  fused_layer_kernel<<<128, 256, 0, stream>>>(H1, W2, b2, nullptr, H2, 0);
  fused_layer_kernel<<<128, 256, 0, stream>>>(H2, W3, b3, u,       Hb, 1);
  fused_layer_kernel<<<128, 256, 0, stream>>>(Hb, Q,  nullptr, nullptr, T, 2);
  qubo_reduce_kernel<<<32, 256, 0, stream>>>(T, Hb, out);
}
